// MultiHeadAttention_50800873177498
// MI455X (gfx1250) — matrix operand order repaired and checked by reading, not compiled, not run on hardware
//
#include <hip/hip_runtime.h>

typedef __attribute__((ext_vector_type(16))) __bf16 v16bf;
typedef __attribute__((ext_vector_type(8)))  __bf16 v8bf;
typedef __attribute__((ext_vector_type(8)))  float  v8f;
typedef __attribute__((ext_vector_type(4)))  float  v4f;

#define WMMA_BF16(a, b, c) \
  __builtin_amdgcn_wmma_f32_16x16x32_bf16(false, (a), false, (b), (short)0, (c), false, false)

constexpr int Bb = 2, Ss = 4096, Dm = 512, Hh = 8, Dk = 64;
constexpr int Mrows = Bb * Ss;  // 8192

__device__ inline v4f ldg4f(const float* p) { return *reinterpret_cast<const v4f*>(p); }
__device__ inline v8bf ldg8bf(const __bf16* p) { return *reinterpret_cast<const v8bf*>(p); }

__device__ inline v8bf cvt8(v4f a, v4f b) {
  v8bf r;
#pragma unroll
  for (int i = 0; i < 4; ++i) { r[i] = (__bf16)a[i]; r[4 + i] = (__bf16)b[i]; }
  return r;
}
__device__ inline v16bf cat8(v8bf lo, v8bf hi) {
  v16bf r;
#pragma unroll
  for (int i = 0; i < 8; ++i) { r[i] = lo[i]; r[8 + i] = hi[i]; }
  return r;
}
__device__ inline v8f zero8f() {
  v8f z = {0.f, 0.f, 0.f, 0.f, 0.f, 0.f, 0.f, 0.f};
  return z;
}

// LDS aperture: generic flat address low 32 bits are the LDS byte offset.
__device__ inline unsigned lds_off_u32(const void* p) {
  return (unsigned)(unsigned long long)p;
}
// Async global->LDS copy, 16 bytes per lane, tracked by ASYNCcnt.
__device__ inline void async_ld_b128(unsigned lds_dst, const void* gsrc) {
  asm volatile("global_load_async_to_lds_b128 %0, %1, off"
               :: "v"(lds_dst), "v"((unsigned long long)(size_t)gsrc)
               : "memory");
}
__device__ inline void wait_async0() { asm volatile("s_wait_asynccnt 0x0" ::: "memory"); }
__device__ inline void wait_async4() { asm volatile("s_wait_asynccnt 0x4" ::: "memory"); }

// ---------------------------------------------------------------------------
// Elementwise fp32 -> bf16 (used to pre-convert the weight matrices).
// ---------------------------------------------------------------------------
__global__ __launch_bounds__(256)
void cvt_kernel(const float* __restrict__ src, __bf16* __restrict__ dst, int n) {
  int i = (blockIdx.x * blockDim.x + threadIdx.x) * 8;
  if (i < n) {
    *reinterpret_cast<v8bf*>(dst + i) = cvt8(ldg4f(src + i), ldg4f(src + i + 4));
  }
}

// ---------------------------------------------------------------------------
// C = X(f32)[M,512] @ Wbf(bf16)[512,512]^T + bias, output bf16.
// vmode 0: out[b,h,s,d] head-major.  vmode 1: out[b,h,d,s] (v transposed).
// Double-buffered k-loop: loads for step k+32 issue before WMMAs of step k.
// ---------------------------------------------------------------------------
struct ProjStage { v4f xa[4]; v8bf wb[4][2]; };

__device__ inline void proj_load(const float* X, const __bf16* Wb, int row,
                                 int colBase, int ml, int h, int kk, ProjStage& s) {
  const float* xp = X + (long)row * Dm + kk + 8 * h;
  s.xa[0] = ldg4f(xp);      s.xa[1] = ldg4f(xp + 4);
  s.xa[2] = ldg4f(xp + 16); s.xa[3] = ldg4f(xp + 20);
#pragma unroll
  for (int t = 0; t < 4; ++t) {
    const __bf16* wp = Wb + (long)(colBase + t * 16 + ml) * Dm + kk + 8 * h;
    s.wb[t][0] = ldg8bf(wp);
    s.wb[t][1] = ldg8bf(wp + 16);
  }
}

__global__ __launch_bounds__(256)
void proj_kernel(const float* __restrict__ X, const __bf16* __restrict__ Wb,
                 const float* __restrict__ bias, __bf16* __restrict__ out, int vmode) {
  const int lane = threadIdx.x & 31;
  const int wave = threadIdx.x >> 5;
  const int ml = lane & 15;
  const int h = lane >> 4;
  const int rowBase = blockIdx.x * 128 + wave * 16;
  const int colBase = blockIdx.y * 64;
  const int row = rowBase + ml;

  v8f acc[4];
#pragma unroll
  for (int t = 0; t < 4; ++t) acc[t] = zero8f();

  ProjStage st[2];
  proj_load(X, Wb, row, colBase, ml, h, 0, st[0]);
#pragma unroll 4
  for (int kk = 0; kk < Dm; kk += 32) {
    const int cur = (kk >> 5) & 1;
    if (kk + 32 < Dm) proj_load(X, Wb, row, colBase, ml, h, kk + 32, st[cur ^ 1]);
    v16bf afrag = cat8(cvt8(st[cur].xa[0], st[cur].xa[1]),
                       cvt8(st[cur].xa[2], st[cur].xa[3]));
#pragma unroll
    for (int t = 0; t < 4; ++t)
      acc[t] = WMMA_BF16(afrag, cat8(st[cur].wb[t][0], st[cur].wb[t][1]), acc[t]);
  }

#pragma unroll
  for (int t = 0; t < 4; ++t) {
    const int col = colBase + t * 16 + ml;
    const float bval = bias[col];
    const int head = col >> 6, d = col & 63;
#pragma unroll
    for (int j = 0; j < 8; ++j) {
      const int r = rowBase + j + 8 * h;
      const int bidx = r >> 12, s = r & (Ss - 1);
      const float v = acc[t][j] + bval;
      long idx;
      if (vmode == 0)
        idx = ((long)((bidx * Hh + head) * Ss + s)) * Dk + d;
      else
        idx = ((long)((bidx * Hh + head) * Dk + d)) * Ss + s;
      out[idx] = (__bf16)v;
    }
  }
}

// ---------------------------------------------------------------------------
// Flash attention with async LDS staging of K/V tiles (double-buffered).
// q,k: [B,H,S,64] bf16;  vT: [B,H,64,S] bf16;  comb: [B,S,512] bf16
// ---------------------------------------------------------------------------
__global__ __launch_bounds__(256)
void flash_kernel(const __bf16* __restrict__ qb, const __bf16* __restrict__ kb_,
                  const __bf16* __restrict__ vtb, const int* __restrict__ mask,
                  __bf16* __restrict__ comb) {
  __shared__ __attribute__((aligned(16))) __bf16 Ktile[2][64 * 64];  // [key][dk]
  __shared__ __attribute__((aligned(16))) __bf16 Vtile[2][64 * 64];  // [dk][key]
  __shared__ __attribute__((aligned(16))) __bf16 Pbuf[8 * 16 * 64];

  const int tid = threadIdx.x;
  const int lane = tid & 31;
  const int wave = tid >> 5;
  const int ml = lane & 15;
  const int h = lane >> 4;
  const int qTiles = Ss / 128;  // 32
  const int bh = blockIdx.x / qTiles;
  const int qt = blockIdx.x % qTiles;
  const int b = bh / Hh, head = bh % Hh;
  const int qRow = qt * 128 + wave * 16;

  const __bf16* qh = qb + (long)bh * Ss * Dk;
  const __bf16* kh = kb_ + (long)bh * Ss * Dk;
  const __bf16* vh = vtb + (long)bh * Dk * Ss;
  const int* mrow = mask + b * Ss;
  __bf16* pw = Pbuf + wave * 16 * 64;

  // Q fragments (d_k = 64 -> 2 chunks of 32), loop invariant
  v16bf aq[2];
#pragma unroll
  for (int c = 0; c < 2; ++c) {
    const __bf16* qp = qh + (long)(qRow + ml) * Dk + c * 32 + 8 * h;
    aq[c] = cat8(ldg8bf(qp), ldg8bf(qp + 16));
  }

  v8f O[4];
  float mrun[8], lrun[8];
#pragma unroll
  for (int t = 0; t < 4; ++t) O[t] = zero8f();
#pragma unroll
  for (int j = 0; j < 8; ++j) { mrun[j] = -3.0e38f; lrun[j] = 0.f; }

  // cooperative async issue of one 64-key tile (2 K-chunks + 2 V-chunks/lane)
  auto issue_tile = [&](int kb0, int buf) {
#pragma unroll
    for (int r = 0; r < 2; ++r) {
      const int chunk = tid + r * 256;            // 512 x 16B per matrix
      const int row = chunk >> 3;                 // key (K) or dk (V)
      const int c8 = (chunk & 7) * 8;
      async_ld_b128(lds_off_u32(&Ktile[buf][chunk * 8]),
                    kh + (long)(kb0 + row) * Dk + c8);
      async_ld_b128(lds_off_u32(&Vtile[buf][chunk * 8]),
                    vh + (long)row * Ss + kb0 + c8);
    }
  };

  issue_tile(0, 0);
  const int nIter = Ss / 64;  // 64
  for (int it = 0; it < nIter; ++it) {
    const int kb0 = it * 64;
    const int cur = it & 1;
    __syncthreads();                       // buf cur^1 free for overwrite
    if (it + 1 < nIter) issue_tile(kb0 + 64, cur ^ 1);
    if (it + 1 < nIter) wait_async4(); else wait_async0();  // tile `it` landed
    __syncthreads();                       // all waves' chunks visible
    const __bf16* Kc = Ktile[cur];
    const __bf16* Vc = Vtile[cur];

    // scores S = q @ k^T  (16 x 64 tile, fp32 accum)
    v8f C[4];
#pragma unroll
    for (int t = 0; t < 4; ++t) {
      C[t] = zero8f();
#pragma unroll
      for (int c = 0; c < 2; ++c) {
        const __bf16* kp = Kc + (t * 16 + ml) * 64 + c * 32 + 8 * h;
        v16bf bk = cat8(*reinterpret_cast<const v8bf*>(kp),
                        *reinterpret_cast<const v8bf*>(kp + 16));
        C[t] = WMMA_BF16(aq[c], bk, C[t]);
      }
    }

    // scale 1/sqrt(64) + key mask (additive -1e9, as in reference)
    float sc[4][8];
#pragma unroll
    for (int t = 0; t < 4; ++t) {
      const float madd = (mrow[kb0 + t * 16 + ml] == 0) ? -1.0e9f : 0.0f;
#pragma unroll
      for (int j = 0; j < 8; ++j) sc[t][j] = C[t][j] * 0.125f + madd;
    }

    // online softmax: row stats via shfl_xor inside 16-lane halves
    float corr[8];
#pragma unroll
    for (int j = 0; j < 8; ++j) {
      float rm = fmaxf(fmaxf(sc[0][j], sc[1][j]), fmaxf(sc[2][j], sc[3][j]));
      rm = fmaxf(rm, __shfl_xor(rm, 1, 32));
      rm = fmaxf(rm, __shfl_xor(rm, 2, 32));
      rm = fmaxf(rm, __shfl_xor(rm, 4, 32));
      rm = fmaxf(rm, __shfl_xor(rm, 8, 32));
      const float mnew = fmaxf(mrun[j], rm);
      corr[j] = __expf(mrun[j] - mnew);
      float rs = 0.f;
#pragma unroll
      for (int t = 0; t < 4; ++t) { sc[t][j] = __expf(sc[t][j] - mnew); rs += sc[t][j]; }
      rs += __shfl_xor(rs, 1, 32);
      rs += __shfl_xor(rs, 2, 32);
      rs += __shfl_xor(rs, 4, 32);
      rs += __shfl_xor(rs, 8, 32);
      lrun[j] = lrun[j] * corr[j] + rs;
      mrun[j] = mnew;
    }
#pragma unroll
    for (int t = 0; t < 4; ++t) {
#pragma unroll
      for (int j = 0; j < 8; ++j) O[t][j] *= corr[j];
    }

    // P (C-layout) -> LDS row-major (per-wave region, DS in-order in-wave)
#pragma unroll
    for (int t = 0; t < 4; ++t) {
#pragma unroll
      for (int j = 0; j < 8; ++j)
        pw[(j + 8 * h) * 64 + t * 16 + ml] = (__bf16)sc[t][j];
    }

    // O += P @ V   (keys in chunks of 32; B operand from V tile in LDS)
#pragma unroll
    for (int c = 0; c < 2; ++c) {
      const __bf16* pp = pw + ml * 64 + c * 32 + 8 * h;
      v16bf ap = cat8(*reinterpret_cast<const v8bf*>(pp),
                      *reinterpret_cast<const v8bf*>(pp + 16));
#pragma unroll
      for (int t = 0; t < 4; ++t) {
        const __bf16* vp = Vc + (t * 16 + ml) * 64 + c * 32 + 8 * h;
        v16bf bv = cat8(*reinterpret_cast<const v8bf*>(vp),
                        *reinterpret_cast<const v8bf*>(vp + 16));
        O[t] = WMMA_BF16(ap, bv, O[t]);
      }
    }
  }

  // normalize and write combined [B, S, H*64] as bf16
#pragma unroll
  for (int j = 0; j < 8; ++j) {
    const float inv = 1.0f / lrun[j];
    const int s = qRow + j + 8 * h;
#pragma unroll
    for (int t = 0; t < 4; ++t) {
      const int col = head * Dk + t * 16 + ml;
      comb[((long)(b * Ss + s)) * Dm + col] = (__bf16)(O[t][j] * inv);
    }
  }
}

// ---------------------------------------------------------------------------
// out(f32) = comb(bf16)[8192,512] @ Wo_bf(bf16)^T + b_o   (double-buffered)
// ---------------------------------------------------------------------------
struct OutStage { v8bf ab[2]; v8bf wb[4][2]; };

__device__ inline void out_load(const __bf16* A, const __bf16* Wb, int row,
                                int colBase, int ml, int h, int kk, OutStage& s) {
  const __bf16* ap = A + (long)row * Dm + kk + 8 * h;
  s.ab[0] = ldg8bf(ap);
  s.ab[1] = ldg8bf(ap + 16);
#pragma unroll
  for (int t = 0; t < 4; ++t) {
    const __bf16* wp = Wb + (long)(colBase + t * 16 + ml) * Dm + kk + 8 * h;
    s.wb[t][0] = ldg8bf(wp);
    s.wb[t][1] = ldg8bf(wp + 16);
  }
}

__global__ __launch_bounds__(256)
void outproj_kernel(const __bf16* __restrict__ A, const __bf16* __restrict__ Wb,
                    const float* __restrict__ bias, float* __restrict__ out) {
  const int lane = threadIdx.x & 31;
  const int wave = threadIdx.x >> 5;
  const int ml = lane & 15;
  const int h = lane >> 4;
  const int rowBase = blockIdx.x * 128 + wave * 16;
  const int colBase = blockIdx.y * 64;
  const int row = rowBase + ml;

  v8f acc[4];
#pragma unroll
  for (int t = 0; t < 4; ++t) acc[t] = zero8f();

  OutStage st[2];
  out_load(A, Wb, row, colBase, ml, h, 0, st[0]);
#pragma unroll 4
  for (int kk = 0; kk < Dm; kk += 32) {
    const int cur = (kk >> 5) & 1;
    if (kk + 32 < Dm) out_load(A, Wb, row, colBase, ml, h, kk + 32, st[cur ^ 1]);
    v16bf afrag = cat8(st[cur].ab[0], st[cur].ab[1]);
#pragma unroll
    for (int t = 0; t < 4; ++t)
      acc[t] = WMMA_BF16(afrag, cat8(st[cur].wb[t][0], st[cur].wb[t][1]), acc[t]);
  }

#pragma unroll
  for (int t = 0; t < 4; ++t) {
    const int col = colBase + t * 16 + ml;
    const float bval = bias[col];
#pragma unroll
    for (int j = 0; j < 8; ++j) {
      const int r = rowBase + j + 8 * h;
      out[(long)r * Dm + col] = acc[t][j] + bval;
    }
  }
}

extern "C" void kernel_launch(void* const* d_in, const int* in_sizes, int n_in,
                              void* d_out, int out_size, void* d_ws, size_t ws_size,
                              hipStream_t stream) {
  (void)in_sizes; (void)n_in; (void)out_size; (void)ws_size;
  const float* Q = (const float*)d_in[0];
  const float* K = (const float*)d_in[1];
  const float* V = (const float*)d_in[2];
  const int* mask = (const int*)d_in[3];
  const float* Wq = (const float*)d_in[4];
  const float* bq = (const float*)d_in[5];
  const float* Wk = (const float*)d_in[6];
  const float* bk = (const float*)d_in[7];
  const float* Wv = (const float*)d_in[8];
  const float* bv = (const float*)d_in[9];
  const float* Wo = (const float*)d_in[10];
  const float* bo = (const float*)d_in[11];

  __bf16* ws = (__bf16*)d_ws;
  const long WE = (long)Dm * Dm;            // 262,144 elems per weight
  const long QE = (long)Bb * Hh * Ss * Dk;  // 4,194,304 elems
  __bf16* wqb = ws;
  __bf16* wkb = ws + WE;
  __bf16* wvb = ws + 2 * WE;
  __bf16* wob = ws + 3 * WE;
  __bf16* qbuf = ws + 4 * WE;
  __bf16* kbuf = qbuf + QE;
  __bf16* vtbuf = qbuf + 2 * QE;
  __bf16* comb = qbuf + 3 * QE;

  dim3 blk(256);
  const int cvtBlocks = (int)(WE / 8 / 256);  // 128
  cvt_kernel<<<cvtBlocks, blk, 0, stream>>>(Wq, wqb, (int)WE);
  cvt_kernel<<<cvtBlocks, blk, 0, stream>>>(Wk, wkb, (int)WE);
  cvt_kernel<<<cvtBlocks, blk, 0, stream>>>(Wv, wvb, (int)WE);
  cvt_kernel<<<cvtBlocks, blk, 0, stream>>>(Wo, wob, (int)WE);

  dim3 g(Mrows / 128, Dm / 64);
  proj_kernel<<<g, blk, 0, stream>>>(Q, wqb, bq, qbuf, 0);
  proj_kernel<<<g, blk, 0, stream>>>(K, wkb, bk, kbuf, 0);
  proj_kernel<<<g, blk, 0, stream>>>(V, wvb, bv, vtbuf, 1);
  flash_kernel<<<Bb * Hh * (Ss / 128), blk, 0, stream>>>(qbuf, kbuf, vtbuf, mask, comb);
  outproj_kernel<<<g, blk, 0, stream>>>(comb, wob, bo, (float*)d_out);
}
